// xLSTM_50706383897184
// MI455X (gfx1250) — matrix operand order repaired and checked by reading, not compiled, not run on hardware
//
#include <hip/hip_runtime.h>
#include <hip/hip_bf16.h>
#include <cstdint>

// ---------------------------------------------------------------------------
// xLSTM block (mLSTM layer0 fwd + sLSTM layer1 fwd/bwd) for MI455X (gfx1250).
// GEMMs: v_wmma_f32_16x16x32_bf16 (wave32), double-buffered LDS staging:
//   A tiles via global_load_async_to_lds_b128 (ASYNCcnt pipeline, per lane)
//   B tiles via TENSOR_LOAD_TO_LDS (TDM descriptor, TENSORcnt, wave 0 only)
// ---------------------------------------------------------------------------

typedef __bf16 bf16;
typedef __attribute__((ext_vector_type(16))) __bf16 v16bf;
typedef __attribute__((ext_vector_type(8)))  __bf16 bf16x8;
typedef __attribute__((ext_vector_type(8)))  float  v8f;
typedef __attribute__((ext_vector_type(4)))  unsigned int u32x4;
typedef __attribute__((ext_vector_type(8)))  int i32x8;
typedef __attribute__((ext_vector_type(4)))  int i32x4;

#define T_DIM   32
#define B_DIM   128
#define D_IN    1024
#define HID     2048
#define NH      8
#define UPS     2730
#define UPS_PAD 2752          // round_up(2730, 32)
#define M_FWD   4096          // T*B
#define M_ALL   8192          // fwd + feature-flipped rows for sLSTM

enum { ACT_NONE = 0, ACT_SIGMOID = 1, ACT_SILU = 2, ACT_GELU = 3 };

static __device__ __forceinline__ bf16 f2bf(float f) {
  union { float f; uint32_t u; } c; c.f = f;
  uint32_t r = c.u + 0x7FFFu + ((c.u >> 16) & 1u);   // round-to-nearest-even
  uint16_t h = (uint16_t)(r >> 16);
  return __builtin_bit_cast(bf16, h);
}

// ---------------------------------------------------------------------------
// TDM: load a 2D tile (tile0 x tile1 rows, 2-byte elements) from global into
// LDS at ldsOff (contiguous rows of tile0 elements). dim0Rem/dim1Rem are the
// remaining tensor extents from the tile start (hardware zero-fills OOB).
// D# layout per CDNA5 ISA ch.8 (group0 128b, group1 256b, groups2/3 unused).
// ---------------------------------------------------------------------------
static __device__ __forceinline__ void tdm_load_2d(
    unsigned ldsOff, const void* gaddr, unsigned dim0Rem, unsigned dim1Rem,
    unsigned tile0, unsigned tile1, unsigned long long stride0 /*elements*/) {
  unsigned long long ga = (unsigned long long)(uintptr_t)gaddr;
  u32x4 g0;
  g0[0] = 1u;                                           // count=1 (valid), user
  g0[1] = ldsOff;                                       // lds_addr [63:32]
  g0[2] = (unsigned)(ga & 0xFFFFFFFFu);                 // global_addr low
  g0[3] = (unsigned)((ga >> 32) & 0x01FFFFFFu) | (2u << 30);  // addr hi | type=2
  i32x8 g1;
  g1[0] = (int)(1u << 16);                              // data_size=1 (2 bytes)
  g1[1] = (int)((dim0Rem & 0xFFFFu) << 16);             // tensor_dim0 [79:48]
  g1[2] = (int)((dim0Rem >> 16) | ((dim1Rem & 0xFFFFu) << 16));  // dim0hi|dim1lo
  g1[3] = (int)((dim1Rem >> 16) | (tile0 << 16));       // dim1 hi | tile_dim0
  g1[4] = (int)tile1;                                   // tile_dim1 (tile_dim2=0)
  g1[5] = (int)(unsigned)(stride0 & 0xFFFFFFFFu);       // tensor_dim0_stride lo
  g1[6] = (int)(unsigned)((stride0 >> 32) & 0xFFFFu);   // stride hi (dim1_stride=0)
  g1[7] = 0;
  i32x4 gz; gz[0] = 0; gz[1] = 0; gz[2] = 0; gz[3] = 0; // groups 2/3: unused dims
  i32x8 gz8;
  for (int i = 0; i < 8; ++i) gz8[i] = 0;               // VADDR4 slot: unused
  __builtin_amdgcn_tensor_load_to_lds(g0, g1, gz, gz, gz8, 0);
}

// ---------------------------------------------------------------------------
// Tiled WMMA GEMM:  C[M,N] = act((A[M,K] * W[K,N] + bias) * scale) (+ resid)
// A: bf16 row-major, row length lda (K padded to mult of 32 = KP)
// Wt: bf16, W transposed: N rows x KP cols (zero padded cols >= K)
// Block tile 128x128, BK=32, 8 waves; wave w computes rows [16w,16w+16) x 128.
// blockIdx.z selects a block-diagonal block (strideAz / strideWz / colOffZ).
// Staging (double buffered): every wave issues 2 async b128 ops per stage for
// its A slice (s_wait_asynccnt 0x2 retires the previous stage); wave 0 issues
// one TDM tensor_load for the whole 128x32 B tile (s_wait_tensorcnt 1);
// the workgroup barrier then publishes the stage to all waves.
// ---------------------------------------------------------------------------
#define BM 128
#define BN 128
#define BK 32

__global__ __launch_bounds__(256) void gemm_wmma_kernel(
    const bf16* __restrict__ A, int lda, long strideAz,
    const bf16* __restrict__ Wt, long strideWz,
    const float* __restrict__ bias, int strideBiasZ,
    const float* __restrict__ resid, int ldres,
    float* __restrict__ C, int ldc,
    bf16* __restrict__ Cb, int ldcb,
    int N, int KP, int colOffZ, float scale, int act) {
  const int z = blockIdx.z;
  const bf16* Ab = A + (long)z * strideAz;
  const bf16* Wb = Wt + (long)z * strideWz;
  const float* biasb = bias ? (bias + (long)z * strideBiasZ) : nullptr;
  const int colOff = z * colOffZ;

  const int rowTile = blockIdx.y * BM;
  const int colTile = blockIdx.x * BN;

  __shared__ alignas(16) bf16 As[2][BM][BK];
  __shared__ alignas(16) bf16 Bs[2][BN][BK];   // Bs[.][n][k] = W[k][n]

  const int tid  = threadIdx.x;
  const int lane = tid & 31;
  const int wv   = tid >> 5;
  const int r    = lane & 15;
  const int hh   = lane >> 4;

  v8f z8; for (int e = 0; e < 8; ++e) z8[e] = 0.f;
  v8f acc[8];
#pragma unroll
  for (int i = 0; i < 8; ++i) acc[i] = z8;

  const int row2 = tid >> 1;            // 0..127
  const int kseg = (tid & 1) * 16;      // 0 or 16
  const long aRowBase = (long)(rowTile + row2) * lda;

  // LDS byte offsets (flat LDS address low 32 bits == wave-relative offset)
  const unsigned ldsA0 = (unsigned)(uintptr_t)&As[0][row2][kseg];
  const unsigned ldsA1 = (unsigned)(uintptr_t)&As[1][row2][kseg];
  const unsigned ldsBt0 = (unsigned)(uintptr_t)&Bs[0][0][0];
  const unsigned ldsBt1 = (unsigned)(uintptr_t)&Bs[1][0][0];
  const unsigned dim1Rem = (unsigned)(N - colTile);     // >= 1 by grid sizing

  // issue one stage: 2 async b128 for A (all waves), 1 TDM for B (wave 0)
  auto issue_stage = [&](int k0, int buf) {
    const bf16* ga = Ab + aRowBase + k0 + kseg;
    unsigned la = buf ? ldsA1 : ldsA0;
    asm volatile("global_load_async_to_lds_b128 %0, %1, off"
                 :: "v"(la), "v"(ga) : "memory");
    asm volatile("global_load_async_to_lds_b128 %0, %1, off offset:16"
                 :: "v"(la), "v"(ga) : "memory");
    if (wv == 0) {
      tdm_load_2d(buf ? ldsBt1 : ldsBt0,
                  Wb + (long)colTile * KP + k0,
                  (unsigned)(KP - k0), dim1Rem, BK, BN,
                  (unsigned long long)KP);
    }
  };

  const int S = KP / BK;
  issue_stage(0, 0);

  for (int s = 0; s < S; ++s) {
    __syncthreads();                       // all reads of buf (s+1)&1 done
    if (s + 1 < S) {
      issue_stage((s + 1) * BK, (s + 1) & 1);
      asm volatile("s_wait_asynccnt 0x2" ::: "memory");   // stage-s A landed
      if (wv == 0) __builtin_amdgcn_s_wait_tensorcnt(1);  // stage-s B landed
    } else {
      asm volatile("s_wait_asynccnt 0x0" ::: "memory");
      if (wv == 0) __builtin_amdgcn_s_wait_tensorcnt(0);
    }
    asm volatile("s_wait_xcnt 0x0" ::: "memory");
    __syncthreads();                       // stage s visible to all waves
    const int buf = s & 1;

    // ---- A fragment: lane holds row wv*16+r; K per ISA 16-bit A layout ----
    const int arow = wv * 16 + r;
    v16bf afrag;
    {
      bf16x8 lo = *(const bf16x8*)&As[buf][arow][hh * 8];
      bf16x8 hi = *(const bf16x8*)&As[buf][arow][16 + hh * 8];
      for (int e = 0; e < 8; ++e) { afrag[e] = lo[e]; afrag[e + 8] = hi[e]; }
    }
#pragma unroll
    for (int nt = 0; nt < 8; ++nt) {
      v16bf bfrag;
      bf16x8 lo = *(const bf16x8*)&Bs[buf][nt * 16 + r][hh * 8];
      bf16x8 hi = *(const bf16x8*)&Bs[buf][nt * 16 + r][16 + hh * 8];
      for (int e = 0; e < 8; ++e) { bfrag[e] = lo[e]; bfrag[e + 8] = hi[e]; }
      acc[nt] = __builtin_amdgcn_wmma_f32_16x16x32_bf16(
          false, afrag, false, bfrag, (short)0, acc[nt], false, false);
    }
  }

  // ---- epilogue: D lane layout: col = r, row = v + 8*hh ----
#pragma unroll
  for (int nt = 0; nt < 8; ++nt) {
    const int ncol = colTile + nt * 16 + r;
    const int nout = colOff + ncol;
#pragma unroll
    for (int v = 0; v < 8; ++v) {
      const long mrow = rowTile + wv * 16 + v + 8 * hh;
      float val = acc[nt][v];
      if (ncol < N) {
        if (biasb) val += biasb[ncol];
        val *= scale;
        if      (act == ACT_SIGMOID) val = 1.f / (1.f + __expf(-val));
        else if (act == ACT_SILU)    val = val / (1.f + __expf(-val));
        else if (act == ACT_GELU)    val = 0.5f * val * (1.f + erff(val * 0.70710678f));
        if (resid) val += resid[mrow * ldres + ncol];
        if (C) C[mrow * ldc + nout] = val;
      } else {
        val = 0.f;
      }
      if (Cb && nout < ldcb) Cb[mrow * ldcb + nout] = f2bf(val);
    }
  }
}

// ---------------------------------------------------------------------------
// Transpose + convert: W[K,N] f32 -> Wt[N,KP] bf16 (zero pad k >= K).
// grid: (ceil(KP/32), ceil(N/32), Z)
// ---------------------------------------------------------------------------
__global__ __launch_bounds__(256) void wconvT_kernel(
    const float* __restrict__ W, bf16* __restrict__ Wt,
    int K, int N, int KP, long wStrideZ, long wtStrideZ) {
  const int z = blockIdx.z;
  const float* Wz = W + (long)z * wStrideZ;
  bf16* Wtz = Wt + (long)z * wtStrideZ;
  __shared__ float tile[32][33];
  const int k0 = blockIdx.x * 32, n0 = blockIdx.y * 32;
  const int tx = threadIdx.x & 31, tg = threadIdx.x >> 5;
  for (int i = tg; i < 32; i += 8) {
    int k = k0 + i, n = n0 + tx;
    tile[i][tx] = (k < K && n < N) ? Wz[(long)k * N + n] : 0.f;
  }
  __syncthreads();
  for (int i = tg; i < 32; i += 8) {
    int n = n0 + i, k = k0 + tx;
    if (n < N && k < KP) Wtz[(long)n * KP + k] = f2bf(tile[tx][i]);
  }
}

__global__ void cvt_bf16_kernel(const float* __restrict__ s, bf16* __restrict__ d, long n) {
  long i = (long)blockIdx.x * 256 + threadIdx.x;
  if (i < n) d[i] = f2bf(s[i]);
}

// ---------------------------------------------------------------------------
// LayerNorm over last dim D; one block per row; optional f32 and bf16 outputs.
// ---------------------------------------------------------------------------
__global__ __launch_bounds__(256) void ln_kernel(
    const float* __restrict__ x, const float* __restrict__ w, const float* __restrict__ b,
    float* __restrict__ yf, bf16* __restrict__ yb, int D) {
  const long row = blockIdx.x;
  const float* xr = x + row * (long)D;
  float s = 0.f, s2 = 0.f;
  for (int j = threadIdx.x; j < D; j += 256) { float v = xr[j]; s += v; s2 += v * v; }
  __shared__ float red0[256], red1[256];
  red0[threadIdx.x] = s; red1[threadIdx.x] = s2; __syncthreads();
  for (int st = 128; st > 0; st >>= 1) {
    if (threadIdx.x < st) { red0[threadIdx.x] += red0[threadIdx.x + st];
                            red1[threadIdx.x] += red1[threadIdx.x + st]; }
    __syncthreads();
  }
  const float mean = red0[0] / D;
  const float var  = red1[0] / D - mean * mean;
  const float rinv = rsqrtf(var + 1e-5f);
  for (int j = threadIdx.x; j < D; j += 256) {
    float v = (xr[j] - mean) * rinv * w[j] + b[j];
    if (yf) yf[row * (long)D + j] = v;
    if (yb) yb[row * (long)D + j] = f2bf(v);
  }
}

// causal depthwise conv (kernel 4, left pad 3) along feature dim + SiLU -> bf16
__global__ void cconv_silu_kernel(const float* __restrict__ x,
                                  const float* __restrict__ cw, const float* __restrict__ cb,
                                  bf16* __restrict__ yb, int D) {
  const long row = blockIdx.y;
  const int j = blockIdx.x * 256 + threadIdx.x;
  if (j >= D) return;
  const float* xr = x + row * (long)D;
  float acc = cb[0];
#pragma unroll
  for (int t = 0; t < 4; ++t) { int jj = j - 3 + t; if (jj >= 0) acc += cw[t] * xr[jj]; }
  float v = acc / (1.f + __expf(-acc));
  yb[row * (long)D + j] = f2bf(v);
}

// mLSTM gates -> c_t (f32), n_t (bf16)
__global__ void mgate_kernel(const float* __restrict__ itf, const float* __restrict__ ftf,
                             const float* __restrict__ vf, const float* __restrict__ kf,
                             const float* __restrict__ mprev, const float* __restrict__ cprev,
                             const float* __restrict__ nprev,
                             float* __restrict__ ct, bf16* __restrict__ nb) {
  long idx = (long)blockIdx.x * 256 + threadIdx.x;
  if (idx >= (long)M_FWD * HID) return;
  long row = idx / HID; int j = (int)(idx % HID);
  int b = (int)(row & (B_DIM - 1));
  long sb = (long)b * HID + j;
  float mp = mprev[sb];
  float iv = itf[idx], fv = ftf[idx];
  float mt = fmaxf(fv + mp, iv);
  float ie = __expf(iv - mt), fe = __expf(fv + mp - mt);
  float kk = kf[idx];
  ct[idx] = fe * cprev[sb] + ie * (vf[idx] * kk);
  nb[idx] = f2bf(fe * nprev[sb] + ie * kk);
}

// ---------------------------------------------------------------------------
// denom[t, j] = max_k | sum_b n[t,b,j] * q[t,b,k] |  via WMMA (K-dim = batch).
// grid: (HID/128 jtiles, T); block 256 (8 waves, each wave a 16-j strip).
// ---------------------------------------------------------------------------
__global__ __launch_bounds__(256) void denom_kernel(
    const bf16* __restrict__ nb, const bf16* __restrict__ qb, float* __restrict__ dn) {
  const int jt = blockIdx.x, t = blockIdx.y;
  const int j0 = jt * 128;
  __shared__ alignas(16) bf16 An[128][128];   // An[j][b]
  __shared__ alignas(16) bf16 Qt[64][128];    // Qt[kcol][b]
  const int tid = threadIdx.x;
  for (int i = tid; i < 128 * 128; i += 256) {
    int b = i >> 7, j = i & 127;
    An[j][b] = nb[((long)t * 128 + b) * HID + j0 + j];
  }
  __syncthreads();
  const int lane = tid & 31, wv = tid >> 5;
  const int r = lane & 15, hh = lane >> 4;
  const int arow = wv * 16 + r;
  v16bf af[4];
#pragma unroll
  for (int ks = 0; ks < 4; ++ks) {
    bf16x8 lo = *(const bf16x8*)&An[arow][ks * 32 + hh * 8];
    bf16x8 hi = *(const bf16x8*)&An[arow][ks * 32 + 16 + hh * 8];
    for (int e = 0; e < 8; ++e) { af[ks][e] = lo[e]; af[ks][e + 8] = hi[e]; }
  }
  v8f z8; for (int e = 0; e < 8; ++e) z8[e] = 0.f;
  float mx[8];
#pragma unroll
  for (int v = 0; v < 8; ++v) mx[v] = 0.f;

  for (int kc = 0; kc < HID; kc += 64) {
    __syncthreads();
    for (int i = tid; i < 64 * 128; i += 256) {
      int b = i >> 6, k = i & 63;
      Qt[k][b] = qb[((long)t * 128 + b) * HID + kc + k];
    }
    __syncthreads();
#pragma unroll
    for (int nt = 0; nt < 4; ++nt) {
      v8f acc = z8;
#pragma unroll
      for (int ks = 0; ks < 4; ++ks) {
        v16bf bfrag;
        bf16x8 lo = *(const bf16x8*)&Qt[nt * 16 + r][ks * 32 + hh * 8];
        bf16x8 hi = *(const bf16x8*)&Qt[nt * 16 + r][ks * 32 + 16 + hh * 8];
        for (int e = 0; e < 8; ++e) { bfrag[e] = lo[e]; bfrag[e + 8] = hi[e]; }
        acc = __builtin_amdgcn_wmma_f32_16x16x32_bf16(
            false, af[ks], false, bfrag, (short)0, acc, false, false);
      }
#pragma unroll
      for (int v = 0; v < 8; ++v) mx[v] = fmaxf(mx[v], fabsf(acc[v]));
    }
  }
#pragma unroll
  for (int m = 1; m < 16; m <<= 1)
#pragma unroll
    for (int v = 0; v < 8; ++v)
      mx[v] = fmaxf(mx[v], __shfl_xor(mx[v], m, 32));
  if (r == 0) {
#pragma unroll
    for (int v = 0; v < 8; ++v)
      dn[(long)t * HID + j0 + wv * 16 + hh * 8 + v] = mx[v];
  }
}

// h_t = o*c*q/denom; GroupNorm(8) + xs; * silu(xr) -> bf16 (block per row)
__global__ __launch_bounds__(256) void mh_gn_kernel(
    const float* __restrict__ of, const float* __restrict__ ct, const float* __restrict__ qf,
    const float* __restrict__ dn, const float* __restrict__ xs, const float* __restrict__ xr,
    const float* __restrict__ gw, const float* __restrict__ gb, bf16* __restrict__ outb) {
  const long row = blockIdx.x;
  const int t = (int)(row / B_DIM);
  const long base = row * HID;
  const int j0 = threadIdx.x * 8;     // warp w covers cols [256w, 256w+256) = group w
  float h[8]; float s = 0.f, s2 = 0.f;
#pragma unroll
  for (int u = 0; u < 8; ++u) {
    int j = j0 + u;
    float hv = of[base + j] * ct[base + j] * qf[base + j] / dn[(long)t * HID + j];
    h[u] = hv; s += hv; s2 += hv * hv;
  }
#pragma unroll
  for (int m = 1; m < 32; m <<= 1) { s += __shfl_xor(s, m, 32); s2 += __shfl_xor(s2, m, 32); }
  const float mean = s * (1.f / 256.f);
  const float var  = s2 * (1.f / 256.f) - mean * mean;
  const float rinv = rsqrtf(var + 1e-5f);
#pragma unroll
  for (int u = 0; u < 8; ++u) {
    int j = j0 + u;
    float g = (h[u] - mean) * rinv * gw[j] + gb[j] + xs[base + j];
    float xv = xr[base + j];
    outb[base + j] = f2bf(g * (xv / (1.f + __expf(-xv))));
  }
}

// build sLSTM input: rows 0..4095 = fwd0, rows 4096..8191 = feature-flipped fwd0
__global__ void build_a_kernel(const float* __restrict__ f0,
                               float* __restrict__ a, bf16* __restrict__ ab) {
  long idx = (long)blockIdx.x * 256 + threadIdx.x;
  if (idx >= (long)M_ALL * D_IN) return;
  long row = idx / D_IN; int j = (int)(idx % D_IN);
  float v = (row < M_FWD) ? f0[row * D_IN + j] : f0[(row - M_FWD) * D_IN + (D_IN - 1 - j)];
  a[idx] = v; ab[idx] = f2bf(v);
}

// sLSTM gates + GroupNorm -> gn bf16; emit h/c/n/m states for t=T-1 fwd rows
__global__ __launch_bounds__(256) void sgate_gn_kernel(
    const float* __restrict__ zW, const float* __restrict__ oW,
    const float* __restrict__ iW, const float* __restrict__ fW,
    const float* __restrict__ Rzh, const float* __restrict__ Roh,
    const float* __restrict__ Rih, const float* __restrict__ Rfh,
    const float* __restrict__ cprev, const float* __restrict__ nprev,
    const float* __restrict__ mprev,
    const float* __restrict__ gw, const float* __restrict__ gb,
    bf16* __restrict__ gnb, float* __restrict__ stateOut) {
  const long row = blockIdx.x;
  const long sfull = (row < M_FWD) ? row : (row - M_FWD);
  const int b = (int)(sfull & (B_DIM - 1));
  const bool emit = (row < M_FWD) && (sfull >= (long)(T_DIM - 1) * B_DIM);
  const long base = row * HID, hb = (long)b * HID;
  const int j0 = threadIdx.x * 8;
  float h[8]; float s = 0.f, s2 = 0.f;
#pragma unroll
  for (int u = 0; u < 8; ++u) {
    int j = j0 + u; long ij = base + j, bj = hb + j;
    float z  = tanhf(zW[ij] + Rzh[bj]);
    float o  = 1.f / (1.f + __expf(-(oW[ij] + Roh[bj])));
    float iv = iW[ij] + Rih[bj];
    float fv = fW[ij] + Rfh[bj];
    float mp = mprev[bj];
    float mt = fmaxf(fv + mp, iv);
    float ie = __expf(iv - mt), fe = __expf(fv + mp - mt);
    float c  = fe * cprev[bj] + ie * z;
    float n  = fe * nprev[bj] + ie;
    float hv = o * c / n;
    if (emit) {
      long so = (long)b * HID + j;
      stateOut[so] = hv;
      stateOut[(long)B_DIM * HID + so] = c;
      stateOut[2L * B_DIM * HID + so] = n;
      stateOut[3L * B_DIM * HID + so] = mt;
    }
    h[u] = hv; s += hv; s2 += hv * hv;
  }
#pragma unroll
  for (int m = 1; m < 32; m <<= 1) { s += __shfl_xor(s, m, 32); s2 += __shfl_xor(s2, m, 32); }
  const float mean = s * (1.f / 256.f);
  const float var  = s2 * (1.f / 256.f) - mean * mean;
  const float rinv = rsqrtf(var + 1e-5f);
#pragma unroll
  for (int u = 0; u < 8; ++u) {
    int j = j0 + u;
    gnb[base + j] = f2bf((h[u] - mean) * rinv * gw[j] + gb[j]);
  }
}

// u = upL_out * gelu(upR_out) -> bf16 (zero pad cols >= UPS)
__global__ void smul_kernel(const float* __restrict__ ul, const float* __restrict__ ur,
                            bf16* __restrict__ ub) {
  long idx = (long)blockIdx.x * 256 + threadIdx.x;
  if (idx >= (long)M_ALL * UPS_PAD) return;
  long row = idx / UPS_PAD; int c = (int)(idx % UPS_PAD);
  float v = (c < UPS) ? ul[row * (long)UPS + c] * ur[row * (long)UPS + c] : 0.f;
  ub[idx] = f2bf(v);
}

// ---------------------------------------------------------------------------
// host side
// ---------------------------------------------------------------------------
static inline void launch_gemm(hipStream_t st,
                               const bf16* A, int lda, long sAz,
                               const bf16* Wt, long sWz,
                               const float* bias, int sBz,
                               const float* resid, int ldres,
                               float* C, int ldc, bf16* Cb, int ldcb,
                               int M, int N, int KP, int colOffZ, int Z,
                               float scale, int act) {
  dim3 grid((N + BN - 1) / BN, M / BM, Z);
  gemm_wmma_kernel<<<grid, 256, 0, st>>>(A, lda, sAz, Wt, sWz, bias, sBz,
                                         resid, ldres, C, ldc, Cb, ldcb,
                                         N, KP, colOffZ, scale, act);
}

static inline void launch_wT(hipStream_t st, const float* W, bf16* Wt,
                             int K, int N, int KP, long wsz, long wtsz, int Z) {
  dim3 grid((KP + 31) / 32, (N + 31) / 32, Z);
  wconvT_kernel<<<grid, 256, 0, st>>>(W, Wt, K, N, KP, wsz, wtsz);
}

extern "C" void kernel_launch(void* const* d_in, const int* in_sizes, int n_in,
                              void* d_out, int out_size, void* d_ws, size_t ws_size,
                              hipStream_t stream) {
  (void)in_sizes; (void)n_in; (void)out_size; (void)ws_size;

  // Inputs: insertion-order recursive flatten of setup_inputs() dict.
  const float* X = (const float*)d_in[0];                       // (T*B, DIN)
  const float *m_lnw = (const float*)d_in[1],  *m_lnb = (const float*)d_in[2];
  const float *m_upLw= (const float*)d_in[3],  *m_upLb= (const float*)d_in[4];
  const float *m_upRw= (const float*)d_in[5],  *m_upRb= (const float*)d_in[6];
  const float *m_cw  = (const float*)d_in[7],  *m_cb  = (const float*)d_in[8];
  const float *m_skw = (const float*)d_in[9],  *m_skb = (const float*)d_in[10];
  const float *m_Wqw = (const float*)d_in[11], *m_Wqb = (const float*)d_in[12];
  const float *m_Wkw = (const float*)d_in[13], *m_Wkb = (const float*)d_in[14];
  const float *m_Wvw = (const float*)d_in[15], *m_Wvb = (const float*)d_in[16];
  const float *m_Wiw = (const float*)d_in[17], *m_Wib = (const float*)d_in[18];
  const float *m_Wfw = (const float*)d_in[19], *m_Wfb = (const float*)d_in[20];
  const float *m_Wow = (const float*)d_in[21], *m_Wob = (const float*)d_in[22];
  const float *m_gnw = (const float*)d_in[23], *m_gnb = (const float*)d_in[24];
  const float *m_dww = (const float*)d_in[25], *m_dwb = (const float*)d_in[26];
  const float *m_cprev=(const float*)d_in[27], *m_nprev=(const float*)d_in[28];
  const float *m_mprev=(const float*)d_in[29];
  const float *s_lnw = (const float*)d_in[30], *s_lnb = (const float*)d_in[31];
  const float *s_cw  = (const float*)d_in[32], *s_cb  = (const float*)d_in[33];
  const float *s_Wzw = (const float*)d_in[34], *s_Wzb = (const float*)d_in[35];
  const float *s_Wiw = (const float*)d_in[36], *s_Wib = (const float*)d_in[37];
  const float *s_Wfw = (const float*)d_in[38], *s_Wfb = (const float*)d_in[39];
  const float *s_Wow = (const float*)d_in[40], *s_Wob = (const float*)d_in[41];
  const float *s_Rzw = (const float*)d_in[42], *s_Rzb = (const float*)d_in[43];
  const float *s_Riw = (const float*)d_in[44], *s_Rib = (const float*)d_in[45];
  const float *s_Rfw = (const float*)d_in[46], *s_Rfb = (const float*)d_in[47];
  const float *s_Row = (const float*)d_in[48], *s_Rob = (const float*)d_in[49];
  const float *s_gnw = (const float*)d_in[50], *s_gnb = (const float*)d_in[51];
  const float *s_upLw= (const float*)d_in[52], *s_upLb= (const float*)d_in[53];
  const float *s_upRw= (const float*)d_in[54], *s_upRb= (const float*)d_in[55];
  const float *s_dww = (const float*)d_in[56], *s_dwb = (const float*)d_in[57];
  const float *s_hprev=(const float*)d_in[58], *s_cprev=(const float*)d_in[59];
  const float *s_nprev=(const float*)d_in[60], *s_mprev=(const float*)d_in[61];

  float* out = (float*)d_out;                                   // (T,B,2*DIN)
  float* stateOut = out + (size_t)T_DIM * B_DIM * 2 * D_IN;     // h,c,n,m tails

  // ---- workspace bump allocator ----
  char* base = (char*)d_ws;
  size_t off = 0;
  auto alloc = [&](size_t nbytes) -> void* {
    void* p = base + off;
    off += (nbytes + 255) & ~(size_t)255;
    return p;
  };

  // persistent: bf16 transposed weights + state copies + fwd0
  bf16* upLt  = (bf16*)alloc((size_t)HID * D_IN * 2);      // 2048 x 1024
  bf16* upRt  = (bf16*)alloc((size_t)HID * D_IN * 2);
  bf16* skipt = (bf16*)alloc((size_t)HID * HID * 2);
  bf16* Wqt   = (bf16*)alloc((size_t)8 * 256 * 256 * 2);
  bf16* Wkt   = (bf16*)alloc((size_t)8 * 256 * 256 * 2);
  bf16* Wvt   = (bf16*)alloc((size_t)8 * 256 * 256 * 2);
  bf16* Wit   = (bf16*)alloc((size_t)HID * HID * 2);
  bf16* Wft   = (bf16*)alloc((size_t)HID * HID * 2);
  bf16* Wot   = (bf16*)alloc((size_t)HID * HID * 2);
  bf16* downt = (bf16*)alloc((size_t)D_IN * HID * 2);      // 1024 x 2048
  bf16* sWzt  = (bf16*)alloc((size_t)8 * 256 * 128 * 2);
  bf16* sWit  = (bf16*)alloc((size_t)8 * 256 * 128 * 2);
  bf16* sWft  = (bf16*)alloc((size_t)8 * 256 * 128 * 2);
  bf16* sWot  = (bf16*)alloc((size_t)8 * 256 * 128 * 2);
  bf16* sRzt  = (bf16*)alloc((size_t)8 * 256 * 256 * 2);
  bf16* sRit  = (bf16*)alloc((size_t)8 * 256 * 256 * 2);
  bf16* sRft  = (bf16*)alloc((size_t)8 * 256 * 256 * 2);
  bf16* sRot  = (bf16*)alloc((size_t)8 * 256 * 256 * 2);
  bf16* supLt = (bf16*)alloc((size_t)UPS * HID * 2);       // 2730 x 2048
  bf16* supRt = (bf16*)alloc((size_t)UPS * HID * 2);
  bf16* sdownt= (bf16*)alloc((size_t)D_IN * UPS_PAD * 2);  // 1024 x 2752
  bf16* h_bf  = (bf16*)alloc((size_t)B_DIM * HID * 2);
  float* fwd0 = (float*)alloc((size_t)M_FWD * D_IN * 4);

  const size_t mark = off;

  // ---------------- weight conversion (bf16, transposed, K-padded) ---------
  launch_wT(stream, m_upLw, upLt, D_IN, HID, D_IN, 0, 0, 1);
  launch_wT(stream, m_upRw, upRt, D_IN, HID, D_IN, 0, 0, 1);
  launch_wT(stream, m_skw,  skipt, HID, HID, HID, 0, 0, 1);
  launch_wT(stream, m_Wqw,  Wqt, 256, 256, 256, 256L*256, 256L*256, 8);
  launch_wT(stream, m_Wkw,  Wkt, 256, 256, 256, 256L*256, 256L*256, 8);
  launch_wT(stream, m_Wvw,  Wvt, 256, 256, 256, 256L*256, 256L*256, 8);
  launch_wT(stream, m_Wiw,  Wit, HID, HID, HID, 0, 0, 1);
  launch_wT(stream, m_Wfw,  Wft, HID, HID, HID, 0, 0, 1);
  launch_wT(stream, m_Wow,  Wot, HID, HID, HID, 0, 0, 1);
  launch_wT(stream, m_dww,  downt, HID, D_IN, HID, 0, 0, 1);
  launch_wT(stream, s_Wzw,  sWzt, 128, 256, 128, 128L*256, 128L*256, 8);
  launch_wT(stream, s_Wiw,  sWit, 128, 256, 128, 128L*256, 128L*256, 8);
  launch_wT(stream, s_Wfw,  sWft, 128, 256, 128, 128L*256, 128L*256, 8);
  launch_wT(stream, s_Wow,  sWot, 128, 256, 128, 128L*256, 128L*256, 8);
  launch_wT(stream, s_Rzw,  sRzt, 256, 256, 256, 256L*256, 256L*256, 8);
  launch_wT(stream, s_Riw,  sRit, 256, 256, 256, 256L*256, 256L*256, 8);
  launch_wT(stream, s_Rfw,  sRft, 256, 256, 256, 256L*256, 256L*256, 8);
  launch_wT(stream, s_Row,  sRot, 256, 256, 256, 256L*256, 256L*256, 8);
  launch_wT(stream, s_upLw, supLt, HID, UPS, HID, 0, 0, 1);
  launch_wT(stream, s_upRw, supRt, HID, UPS, HID, 0, 0, 1);
  launch_wT(stream, s_dww,  sdownt, UPS, D_IN, UPS_PAD, 0, 0, 1);
  cvt_bf16_kernel<<<(B_DIM * HID + 255) / 256, 256, 0, stream>>>(
      s_hprev, h_bf, (long)B_DIM * HID);

  // ================= layer0: mLSTM (fwd rows only; bwd is dead) ============
  {
    bf16*  xn_bf = (bf16*) alloc((size_t)M_FWD * D_IN * 2);
    float* xl    = (float*)alloc((size_t)M_FWD * HID * 4);
    bf16*  xl_bf = (bf16*) alloc((size_t)M_FWD * HID * 2);
    float* xr    = (float*)alloc((size_t)M_FWD * HID * 4);
    bf16*  xc_bf = (bf16*) alloc((size_t)M_FWD * HID * 2);
    float* xs    = (float*)alloc((size_t)M_FWD * HID * 4);
    float* qf    = (float*)alloc((size_t)M_FWD * HID * 4);
    bf16*  q_bf  = (bf16*) alloc((size_t)M_FWD * HID * 2);
    float* kf    = (float*)alloc((size_t)M_FWD * HID * 4);
    float* vf    = (float*)alloc((size_t)M_FWD * HID * 4);
    float* itf   = (float*)alloc((size_t)M_FWD * HID * 4);
    float* ftf   = (float*)alloc((size_t)M_FWD * HID * 4);
    float* of    = (float*)alloc((size_t)M_FWD * HID * 4);
    float* ct    = (float*)alloc((size_t)M_FWD * HID * 4);
    bf16*  n_bf  = (bf16*) alloc((size_t)M_FWD * HID * 2);
    float* dn    = (float*)alloc((size_t)T_DIM * HID * 4);
    bf16*  pd_bf = (bf16*) alloc((size_t)M_FWD * HID * 2);

    ln_kernel<<<M_FWD, 256, 0, stream>>>(X, m_lnw, m_lnb, nullptr, xn_bf, D_IN);
    launch_gemm(stream, xn_bf, D_IN, 0, upLt, 0, m_upLb, 0, nullptr, 0,
                xl, HID, xl_bf, HID, M_FWD, HID, D_IN, 0, 1, 1.f, ACT_NONE);
    launch_gemm(stream, xn_bf, D_IN, 0, upRt, 0, m_upRb, 0, nullptr, 0,
                xr, HID, nullptr, 0, M_FWD, HID, D_IN, 0, 1, 1.f, ACT_NONE);
    cconv_silu_kernel<<<dim3((HID + 255) / 256, M_FWD), 256, 0, stream>>>(
        xl, m_cw, m_cb, xc_bf, HID);
    launch_gemm(stream, xc_bf, HID, 0, skipt, 0, m_skb, 0, nullptr, 0,
                xs, HID, nullptr, 0, M_FWD, HID, HID, 0, 1, 1.f, ACT_NONE);
    launch_gemm(stream, xc_bf, HID, 256, Wqt, 256L*256, m_Wqb, 256, nullptr, 0,
                qf, HID, q_bf, HID, M_FWD, 256, 256, 256, 8, 1.f, ACT_NONE);
    launch_gemm(stream, xc_bf, HID, 256, Wkt, 256L*256, m_Wkb, 256, nullptr, 0,
                kf, HID, nullptr, 0, M_FWD, 256, 256, 256, 8, 0.0625f, ACT_NONE);
    launch_gemm(stream, xl_bf, HID, 256, Wvt, 256L*256, m_Wvb, 256, nullptr, 0,
                vf, HID, nullptr, 0, M_FWD, 256, 256, 256, 8, 1.f, ACT_NONE);
    launch_gemm(stream, xc_bf, HID, 0, Wit, 0, m_Wib, 0, nullptr, 0,
                itf, HID, nullptr, 0, M_FWD, HID, HID, 0, 1, 1.f, ACT_NONE);
    launch_gemm(stream, xc_bf, HID, 0, Wft, 0, m_Wfb, 0, nullptr, 0,
                ftf, HID, nullptr, 0, M_FWD, HID, HID, 0, 1, 1.f, ACT_NONE);
    launch_gemm(stream, xl_bf, HID, 0, Wot, 0, m_Wob, 0, nullptr, 0,
                of, HID, nullptr, 0, M_FWD, HID, HID, 0, 1, 1.f, ACT_SIGMOID);
    mgate_kernel<<<(M_FWD * HID) / 256, 256, 0, stream>>>(
        itf, ftf, vf, kf, m_mprev, m_cprev, m_nprev, ct, n_bf);
    denom_kernel<<<dim3(HID / 128, T_DIM), 256, 0, stream>>>(n_bf, q_bf, dn);
    mh_gn_kernel<<<M_FWD, 256, 0, stream>>>(of, ct, qf, dn, xs, xr,
                                            m_gnw, m_gnb, pd_bf);
    launch_gemm(stream, pd_bf, HID, 0, downt, 0, m_dwb, 0, X, D_IN,
                fwd0, D_IN, nullptr, 0, M_FWD, D_IN, HID, 0, 1, 1.f, ACT_NONE);
  }

  // ================= layer1: sLSTM on [fwd0 ; flip(fwd0)] ==================
  off = mark;   // overlay: reuse the mLSTM activation region
  {
    float* a     = (float*)alloc((size_t)M_ALL * D_IN * 4);
    bf16*  a_bf  = (bf16*) alloc((size_t)M_ALL * D_IN * 2);
    float* sxn   = (float*)alloc((size_t)M_ALL * D_IN * 4);
    bf16*  sxc   = (bf16*) alloc((size_t)M_ALL * D_IN * 2);
    float* Rzh   = (float*)alloc((size_t)B_DIM * HID * 4);
    float* Rih   = (float*)alloc((size_t)B_DIM * HID * 4);
    float* Rfh   = (float*)alloc((size_t)B_DIM * HID * 4);
    float* Roh   = (float*)alloc((size_t)B_DIM * HID * 4);
    float* zW    = (float*)alloc((size_t)M_ALL * HID * 4);
    float* oW    = (float*)alloc((size_t)M_ALL * HID * 4);
    float* iW    = (float*)alloc((size_t)M_ALL * HID * 4);
    float* fW    = (float*)alloc((size_t)M_ALL * HID * 4);
    bf16*  gn_bf = (bf16*) alloc((size_t)M_ALL * HID * 2);
    float* ul    = (float*)alloc((size_t)M_ALL * UPS * 4);
    float* ur    = (float*)alloc((size_t)M_ALL * UPS * 4);
    bf16*  u_bf  = (bf16*) alloc((size_t)M_ALL * UPS_PAD * 2);

    build_a_kernel<<<(M_ALL * D_IN) / 256, 256, 0, stream>>>(fwd0, a, a_bf);
    ln_kernel<<<M_ALL, 256, 0, stream>>>(a, s_lnw, s_lnb, sxn, nullptr, D_IN);
    cconv_silu_kernel<<<dim3((D_IN + 255) / 256, M_ALL), 256, 0, stream>>>(
        sxn, s_cw, s_cb, sxc, D_IN);
    // recurrent terms from constant h_prev (128 rows, broadcast by b)
    launch_gemm(stream, h_bf, HID, 256, sRzt, 256L*256, s_Rzb, 256, nullptr, 0,
                Rzh, HID, nullptr, 0, B_DIM, 256, 256, 256, 8, 1.f, ACT_NONE);
    launch_gemm(stream, h_bf, HID, 256, sRit, 256L*256, s_Rib, 256, nullptr, 0,
                Rih, HID, nullptr, 0, B_DIM, 256, 256, 256, 8, 1.f, ACT_NONE);
    launch_gemm(stream, h_bf, HID, 256, sRft, 256L*256, s_Rfb, 256, nullptr, 0,
                Rfh, HID, nullptr, 0, B_DIM, 256, 256, 256, 8, 1.f, ACT_NONE);
    launch_gemm(stream, h_bf, HID, 256, sRot, 256L*256, s_Rob, 256, nullptr, 0,
                Roh, HID, nullptr, 0, B_DIM, 256, 256, 256, 8, 1.f, ACT_NONE);
    // input projections (block-diagonal, 8 x (128 -> 256))
    launch_gemm(stream, a_bf, D_IN, 128, sWzt, 128L*256, s_Wzb, 256, nullptr, 0,
                zW, HID, nullptr, 0, M_ALL, 256, 128, 256, 8, 1.f, ACT_NONE);
    launch_gemm(stream, a_bf, D_IN, 128, sWot, 128L*256, s_Wob, 256, nullptr, 0,
                oW, HID, nullptr, 0, M_ALL, 256, 128, 256, 8, 1.f, ACT_NONE);
    launch_gemm(stream, sxc, D_IN, 128, sWit, 128L*256, s_Wib, 256, nullptr, 0,
                iW, HID, nullptr, 0, M_ALL, 256, 128, 256, 8, 1.f, ACT_NONE);
    launch_gemm(stream, sxc, D_IN, 128, sWft, 128L*256, s_Wfb, 256, nullptr, 0,
                fW, HID, nullptr, 0, M_ALL, 256, 128, 256, 8, 1.f, ACT_NONE);
    sgate_gn_kernel<<<M_ALL, 256, 0, stream>>>(
        zW, oW, iW, fW, Rzh, Roh, Rih, Rfh,
        s_cprev, s_nprev, s_mprev, s_gnw, s_gnb, gn_bf, stateOut);
    launch_gemm(stream, gn_bf, HID, 0, supLt, 0, s_upLb, 0, nullptr, 0,
                ul, UPS, nullptr, 0, M_ALL, UPS, HID, 0, 1, 1.f, ACT_NONE);
    launch_gemm(stream, gn_bf, HID, 0, supRt, 0, s_upRb, 0, nullptr, 0,
                ur, UPS, nullptr, 0, M_ALL, UPS, HID, 0, 1, 1.f, ACT_GELU);
    smul_kernel<<<((long)M_ALL * UPS_PAD + 255) / 256, 256, 0, stream>>>(ul, ur, u_bf);
    // down-projection + residual, written straight into d_out halves
    launch_gemm(stream, u_bf, UPS_PAD, 0, sdownt, 0, s_dwb, 0, a, D_IN,
                out, 2 * D_IN, nullptr, 0, M_FWD, D_IN, UPS_PAD, 0, 1, 1.f, ACT_NONE);
    launch_gemm(stream, u_bf + (size_t)M_FWD * UPS_PAD, UPS_PAD, 0, sdownt, 0,
                s_dwb, 0, a + (size_t)M_FWD * D_IN, D_IN,
                out + D_IN, 2 * D_IN, nullptr, 0, M_FWD, D_IN, UPS_PAD, 0, 1,
                1.f, ACT_NONE);
  }
}
